// SparseNetwork_16827681865729
// MI455X (gfx1250) — matrix operand order repaired and checked by reading, not compiled, not run on hardware
//
#include <hip/hip_runtime.h>
#include <stdint.h>

typedef __attribute__((ext_vector_type(16))) __bf16 bf16x16;
typedef __attribute__((ext_vector_type(8)))  float  f32x8;

#define N_DIM 2048
#define B_DIM 16384
#define BM 256            // batch rows per block
#define BN 128            // output cols per block
#define BK 32
#define LDSS 40           // 32 + 8 pad, in bf16 elements (80B row stride, 16B aligned)

__device__ __forceinline__ ushort f2bf(float f) {
    // round-to-nearest-even fp32 -> bf16
    uint32_t u = __builtin_bit_cast(uint32_t, f);
    uint32_t r = u + 0x7FFFu + ((u >> 16) & 1u);
    return (ushort)(r >> 16);
}

// Low 32 bits of a generic pointer to __shared__ = LDS byte offset (ISA aperture rule).
__device__ __forceinline__ uint32_t lds_off(const void* p) {
    return (uint32_t)(uintptr_t)p;
}

// gfx1250 async global->LDS copy, 16B per lane, tracked by ASYNCcnt.
__device__ __forceinline__ void async_b128(uint32_t lds_addr, const void* g) {
    asm volatile("global_load_async_to_lds_b128 %0, %1, off"
                 :: "v"(lds_addr), "v"(g) : "memory");
}
__device__ __forceinline__ void wait_async0() {
    asm volatile("s_wait_asynccnt 0x0" ::: "memory");
}

// Apply triangular mask and convert W (fp32 [N,N], row = out index m, col = k)
// to bf16. upper=1 keeps k>=m (triu), upper=0 keeps k<=m (tril).
__global__ __launch_bounds__(256) void tri_mask_bf16(const float* __restrict__ W,
                                                     ushort* __restrict__ Wb,
                                                     int upper) {
    int idx  = blockIdx.x * 256 + threadIdx.x;   // one thread per 4 elements
    int base = idx * 4;
    int m  = base >> 11;          // /2048
    int k0 = base & (N_DIM - 1);
    float4 w = *(const float4*)(W + base);
    ushort4 o;
    o.x = (upper ? (k0 + 0 >= m) : (k0 + 0 <= m)) ? f2bf(w.x) : (ushort)0;
    o.y = (upper ? (k0 + 1 >= m) : (k0 + 1 <= m)) ? f2bf(w.y) : (ushort)0;
    o.z = (upper ? (k0 + 2 >= m) : (k0 + 2 <= m)) ? f2bf(w.z) : (ushort)0;
    o.w = (upper ? (k0 + 3 >= m) : (k0 + 3 <= m)) ? f2bf(w.w) : (ushort)0;
    *(ushort4*)(Wb + base) = o;
}

// Y = epilogue(A @ Wb^T + bias).
//   A_IS_F32:  A is fp32 (convert to bf16 on the fly) else bf16 (async to LDS).
//   UPPER:     triangular K skip. upper: k in [m0, N); lower: k in [0, m0+BN).
//   RELU_BF16: relu + bf16 store (intermediate H), else plain fp32 store.
template <bool A_IS_F32, bool UPPER, bool RELU_BF16>
__global__ __launch_bounds__(256)
void gemm_tri_bf16(const void* __restrict__ Aptr, const ushort* __restrict__ Wb,
                   const float* __restrict__ bias, void* __restrict__ Out) {
    __shared__ __align__(16) ushort lds_a[2][BM * LDSS];   // 2 x 20 KB
    __shared__ __align__(16) ushort lds_w[2][BN * LDSS];   // 2 x 10 KB

    const int tid  = threadIdx.x;
    const int lane = tid & 31;
    const int wave = tid >> 5;        // 0..7
    const int wm   = wave & 3;        // 4 waves over M (64 rows each)
    const int wn   = wave >> 2;       // 2 waves over N (64 cols each)
    const int l16  = lane & 15;
    const int lh   = lane >> 4;       // lane half

    const int brow0 = blockIdx.y * BM;       // batch rows
    const int m0    = blockIdx.x * BN;       // output cols (= W rows)

    const int k_begin = UPPER ? m0 : 0;
    const int k_end   = UPPER ? N_DIM : (m0 + BN);
    const int nsteps  = (k_end - k_begin) / BK;

    f32x8 acc[4][4];
#pragma unroll
    for (int mi = 0; mi < 4; ++mi)
#pragma unroll
        for (int ni = 0; ni < 4; ++ni)
#pragma unroll
            for (int r = 0; r < 8; ++r) acc[mi][ni][r] = 0.0f;

    // ---- tile staging (A: fp32 convert via VALU, or bf16 async; W: bf16 async) ----
    auto load_tile = [&](int buf, int k0) {
        if (A_IS_F32) {
            const float* X = (const float*)Aptr;
#pragma unroll
            for (int i = 0; i < 8; ++i) {            // 2048 float4 / 256 thr
                int f   = tid + i * 256;
                int row = f >> 3;                    // 8 float4 per row
                int cg  = f & 7;
                float4 v = *(const float4*)(X + (size_t)(brow0 + row) * N_DIM + k0 + cg * 4);
                uint2 p;
                p.x = (uint)f2bf(v.x) | ((uint)f2bf(v.y) << 16);
                p.y = (uint)f2bf(v.z) | ((uint)f2bf(v.w) << 16);
                *(uint2*)(&lds_a[buf][row * LDSS + cg * 4]) = p;
            }
        } else {
            const ushort* H = (const ushort*)Aptr;
#pragma unroll
            for (int i = 0; i < 4; ++i) {            // 1024 b128 / 256 thr
                int f   = tid + i * 256;
                int row = f >> 2;                    // 4 b128 per row
                int cg  = f & 3;
                async_b128(lds_off(&lds_a[buf][row * LDSS + cg * 8]),
                           H + (size_t)(brow0 + row) * N_DIM + k0 + cg * 8);
            }
        }
#pragma unroll
        for (int i = 0; i < 2; ++i) {                // 512 b128 / 256 thr
            int f   = tid + i * 256;
            int row = f >> 2;
            int cg  = f & 3;
            async_b128(lds_off(&lds_w[buf][row * LDSS + cg * 8]),
                       Wb + (size_t)(m0 + row) * N_DIM + k0 + cg * 8);
        }
    };

    // prologue: stage first tile
    load_tile(0, k_begin);
    wait_async0();
    __syncthreads();

    for (int s = 0; s < nsteps; ++s) {
        const int cur = s & 1;
        if (s + 1 < nsteps) load_tile(cur ^ 1, k_begin + (s + 1) * BK);

        // ---- fragments per ISA 7.12.2 layouts ----
        union FragU { bf16x16 v; uint4 q[2]; };
        FragU af[4], br[4];
#pragma unroll
        for (int mi = 0; mi < 4; ++mi) {             // A: M = base+l16, K chunks h*8 / 16+h*8
            const ushort* p = &lds_a[cur][(wm * 64 + mi * 16 + l16) * LDSS];
            af[mi].q[0] = *(const uint4*)(p + lh * 8);
            af[mi].q[1] = *(const uint4*)(p + 16 + lh * 8);
        }
#pragma unroll
        for (int ni = 0; ni < 4; ++ni) {             // B: N = base+l16, K chunks h*8 / 16+h*8
            const ushort* p = &lds_w[cur][(wn * 64 + ni * 16 + l16) * LDSS];
            br[ni].q[0] = *(const uint4*)(p + lh * 8);
            br[ni].q[1] = *(const uint4*)(p + 16 + lh * 8);
        }

#pragma unroll
        for (int mi = 0; mi < 4; ++mi)
#pragma unroll
            for (int ni = 0; ni < 4; ++ni)
                acc[mi][ni] = __builtin_amdgcn_wmma_f32_16x16x32_bf16(
                    false, af[mi].v, false, br[ni].v,
                    (short)0, acc[mi][ni], false, false);

        wait_async0();        // drain this wave's async writes into the next buffer
        __syncthreads();      // everyone done reading cur + writing next
    }

    // ---- epilogue: bias (+relu) ; C/D layout: M = r + 8*lh, N = l16 ----
#pragma unroll
    for (int ni = 0; ni < 4; ++ni) {
        int col  = m0 + wn * 64 + ni * 16 + l16;
        float bv = bias[col];
#pragma unroll
        for (int mi = 0; mi < 4; ++mi) {
#pragma unroll
            for (int r = 0; r < 8; ++r) {
                int rowg = brow0 + wm * 64 + mi * 16 + r + 8 * lh;
                float v  = acc[mi][ni][r] + bv;
                if (RELU_BF16) {
                    v = v > 0.0f ? v : 0.0f;
                    ((ushort*)Out)[(size_t)rowg * N_DIM + col] = f2bf(v);
                } else {
                    ((float*)Out)[(size_t)rowg * N_DIM + col] = v;
                }
            }
        }
    }
}

extern "C" void kernel_launch(void* const* d_in, const int* in_sizes, int n_in,
                              void* d_out, int out_size, void* d_ws, size_t ws_size,
                              hipStream_t stream) {
    const float* x  = (const float*)d_in[0];
    const float* Wu = (const float*)d_in[1];
    const float* bu = (const float*)d_in[2];
    const float* Wl = (const float*)d_in[3];
    const float* bl = (const float*)d_in[4];
    float* out = (float*)d_out;

    // ws layout: Wu_bf16 (8MB) | Wl_bf16 (8MB) | H_bf16 (64MB)
    ushort* Wu_b = (ushort*)d_ws;
    ushort* Wl_b = Wu_b + (size_t)N_DIM * N_DIM;
    ushort* Hb   = Wl_b + (size_t)N_DIM * N_DIM;

    dim3 cgrd((N_DIM * (size_t)N_DIM / 4) / 256);
    tri_mask_bf16<<<cgrd, 256, 0, stream>>>(Wu, Wu_b, 1);
    tri_mask_bf16<<<cgrd, 256, 0, stream>>>(Wl, Wl_b, 0);

    dim3 grd(N_DIM / BN, B_DIM / BM);   // 16 x 64 blocks
    // H = relu(x @ triu(Wu)^T + bu), K starts at m0 (triu skip)
    gemm_tri_bf16<true, true, true><<<grd, 256, 0, stream>>>(x, Wu_b, bu, Hb);
    // Y = H @ tril(Wl)^T + bl, K ends at m0+BN (tril skip)
    gemm_tri_bf16<false, false, false><<<grd, 256, 0, stream>>>(Hb, Wl_b, bl, out);
}
